// GGNN_3418793967874
// MI455X (gfx1250) — matrix operand order repaired and checked by reading, not compiled, not run on hardware
//
#include <hip/hip_runtime.h>
#include <hip/hip_bf16.h>

typedef __attribute__((ext_vector_type(16))) _Float16 v16h;
typedef __attribute__((ext_vector_type(8)))  _Float16 v8h;
typedef __attribute__((ext_vector_type(8)))  float    v8f;

#define WAVES 4

// ---------------- WMMA helpers ----------------

__device__ __forceinline__ v8f wmma16(v16h a, v16h b, v8f c) {
  // D = A(16x32 f16) * B(32x16 f16) + C(16x16 f32)
  return __builtin_amdgcn_wmma_f32_16x16x32_f16(false, a, false, b, (short)0, c,
                                                false, false);
}

// A fragment from an LDS row-major f16 buffer [16][stride]; lane = m + 16*hi.
// Lane (m,hi) holds row m, K = {32kt+8hi .. +7} U {32kt+16+8hi .. +7}
__device__ __forceinline__ v16h load_A_lds(const _Float16* buf, int stride,
                                           int m, int hi, int kt) {
  const _Float16* p = buf + m * stride + kt * 32 + 8 * hi;
  v8h lo = *(const v8h*)p;
  v8h hh = *(const v8h*)(p + 16);
  v16h a;
#pragma unroll
  for (int i = 0; i < 8; ++i) { a[i] = lo[i]; a[8 + i] = hh[i]; }
  return a;
}

// A fragment directly from a global f32 row (row-major [.,K]); converts to f16.
__device__ __forceinline__ v16h load_A_g(const float* rowbase, int kt, int hi) {
  const float* p = rowbase + kt * 32 + 8 * hi;
  v16h a;
#pragma unroll
  for (int i = 0; i < 8; ++i) a[i] = (_Float16)p[i];
#pragma unroll
  for (int i = 0; i < 8; ++i) a[8 + i] = (_Float16)p[16 + i];
  return a;
}

// B fragment from global f16 weights in K-major layout Wt[k][n], n-width Nw.
// Lane (n,hi) holds column n, K = {32kt+8hi .. +7} U {32kt+16+8hi .. +7}
__device__ __forceinline__ v16h load_B(const _Float16* Wt, int Nw, int lane,
                                       int kt, int nt) {
  const _Float16* p = Wt + (size_t)(kt * 32 + 8 * (lane >> 4)) * Nw + nt * 16 + (lane & 15);
  v16h b;
#pragma unroll
  for (int i = 0; i < 8; ++i) { b[i] = p[(size_t)i * Nw]; b[8 + i] = p[(size_t)(16 + i) * Nw]; }
  return b;
}

// ---------------- prep kernels ----------------

__global__ void k_zero(float* p, int n) {
  int i = blockIdx.x * blockDim.x + threadIdx.x;
  if (i < n) p[i] = 0.f;
}

// Wt[k][n] = (k < inDim) ? W[n][k] : 0 ; Wt is [Kpad][outDim] f16
__global__ void k_transpose_cvt(const float* __restrict__ W,
                                _Float16* __restrict__ Wt,
                                int outDim, int inDim, int Kpad) {
  int idx = blockIdx.x * blockDim.x + threadIdx.x;
  if (idx >= Kpad * outDim) return;
  int k = idx / outDim, n = idx % outDim;
  Wt[(size_t)k * outDim + n] =
      (k < inDim) ? (_Float16)W[(size_t)n * inDim + k] : (_Float16)0.f;
}

__global__ void k_edge_feat(const float* __restrict__ J, const float* __restrict__ b,
                            const int* __restrict__ row, const int* __restrict__ col,
                            float* __restrict__ ef, int E, int N) {
  int e = blockIdx.x * blockDim.x + threadIdx.x;
  if (e >= E) return;
  int r = row[e], c = col[e];
  ef[e * 4 + 0] = b[r];
  ef[e * 4 + 1] = b[c];
  ef[e * 4 + 2] = J[(size_t)r * N + c];
  ef[e * 4 + 3] = J[(size_t)c * N + r];
}

// ---------------- edge message MLP (the hot kernel) ----------------
// One wave = 16 edges. 3-layer MLP: 160(pad of 132) ->128 ->128 ->64.
__global__ __launch_bounds__(128) void k_edge_mlp(
    const float* __restrict__ h, const float* __restrict__ ef,
    const int* __restrict__ row, const int* __restrict__ col,
    const _Float16* __restrict__ W1t, const float* __restrict__ b1,
    const _Float16* __restrict__ W2t, const float* __restrict__ b2,
    const _Float16* __restrict__ W3t, const float* __restrict__ b3,
    float* __restrict__ x, int E, int nTiles) {
  __shared__ _Float16 bufA[WAVES][16 * 160];
  __shared__ _Float16 bufB[WAVES][16 * 128];

  const int wave = threadIdx.x >> 5;
  const int lane = threadIdx.x & 31;
  const int tile = blockIdx.x * WAVES + wave;
  if (tile >= nTiles) return;                 // wave-uniform
  const int eBase = tile * 16;
  const int m  = lane & 15;
  const int hi = lane >> 4;
  const int nl = lane & 15;

  // ---- stage inputs: lane (m,hf): hf=0 -> h[row], hf=1 -> h[col] + feat/pad
  // packed: cvt 8 floats -> one v8h -> one ds_store_b128
  {
    int e = eBase + m;
    if (e >= E) e = E - 1;
    _Float16* dst = &bufA[wave][m * 160 + hi * 64];
    int node = hi ? col[e] : row[e];
    const float* src = h + (size_t)node * 64;
#pragma unroll
    for (int i = 0; i < 64; i += 8) {
      float4 va = *(const float4*)(src + i);
      float4 vb = *(const float4*)(src + i + 4);
      v8h t;
      t[0] = (_Float16)va.x; t[1] = (_Float16)va.y;
      t[2] = (_Float16)va.z; t[3] = (_Float16)va.w;
      t[4] = (_Float16)vb.x; t[5] = (_Float16)vb.y;
      t[6] = (_Float16)vb.z; t[7] = (_Float16)vb.w;
      *(v8h*)(dst + i) = t;
    }
    if (hi) {
      _Float16* pad = &bufA[wave][m * 160 + 128];  // 16B aligned
      float4 fv = *(const float4*)(ef + (size_t)e * 4);
      v8h t = {};
      t[0] = (_Float16)fv.x; t[1] = (_Float16)fv.y;
      t[2] = (_Float16)fv.z; t[3] = (_Float16)fv.w;
      *(v8h*)(pad + 0) = t;
      v8h z = {};
      *(v8h*)(pad + 8)  = z;
      *(v8h*)(pad + 16) = z;
      *(v8h*)(pad + 24) = z;
    }
  }
  // same-wave LDS RAW: in-order DS + compiler waitcnt, no barrier needed

  const _Float16* A1 = &bufA[wave][0];   // stride 160
  _Float16*       A2 = &bufB[wave][0];   // stride 128
  _Float16*       A3 = &bufA[wave][0];   // reused, stride 128

  // ---- layer 1: K=160 (5 kt), N=128 (8 nt), ReLU
#pragma unroll
  for (int nt = 0; nt < 8; ++nt) {
    v8f acc = {};
#pragma unroll
    for (int kt = 0; kt < 5; ++kt)
      acc = wmma16(load_A_lds(A1, 160, m, hi, kt), load_B(W1t, 128, lane, kt, nt), acc);
    float bias = b1[nt * 16 + nl];
#pragma unroll
    for (int v = 0; v < 8; ++v) {
      float t = acc[v] + bias;
      A2[(v + 8 * hi) * 128 + nt * 16 + nl] = (_Float16)(t > 0.f ? t : 0.f);
    }
  }
  // ---- layer 2: K=128 (4 kt), N=128 (8 nt), ReLU
#pragma unroll
  for (int nt = 0; nt < 8; ++nt) {
    v8f acc = {};
#pragma unroll
    for (int kt = 0; kt < 4; ++kt)
      acc = wmma16(load_A_lds(A2, 128, m, hi, kt), load_B(W2t, 128, lane, kt, nt), acc);
    float bias = b2[nt * 16 + nl];
#pragma unroll
    for (int v = 0; v < 8; ++v) {
      float t = acc[v] + bias;
      A3[(v + 8 * hi) * 128 + nt * 16 + nl] = (_Float16)(t > 0.f ? t : 0.f);
    }
  }
  // ---- layer 3: K=128 (4 kt), N=64 (4 nt) -> scatter-add into x[col]
  int cols8[8];
#pragma unroll
  for (int v = 0; v < 8; ++v) {
    int ee = eBase + v + 8 * hi;
    cols8[v] = (ee < E) ? col[ee] : -1;
  }
#pragma unroll
  for (int nt = 0; nt < 4; ++nt) {
    v8f acc = {};
#pragma unroll
    for (int kt = 0; kt < 4; ++kt)
      acc = wmma16(load_A_lds(A3, 128, m, hi, kt), load_B(W3t, 64, lane, kt, nt), acc);
    float bias = b3[nt * 16 + nl];
#pragma unroll
    for (int v = 0; v < 8; ++v) {
      if (cols8[v] >= 0)
        unsafeAtomicAdd(&x[(size_t)cols8[v] * 64 + nt * 16 + nl], acc[v] + bias);
    }
  }
}

// ---------------- GRU update ----------------
// gi = x@Wih^T, gh = h@Whh^T (both [16 x 192] per tile); gate math; h in place.
__global__ __launch_bounds__(128) void k_gru(
    const float* __restrict__ x, float* __restrict__ h,
    const _Float16* __restrict__ Wiht, const float* __restrict__ b_ih,
    const _Float16* __restrict__ Whht, const float* __restrict__ b_hh,
    int nTiles, int N) {
  const int wave = threadIdx.x >> 5;
  const int lane = threadIdx.x & 31;
  const int tile = blockIdx.x * WAVES + wave;
  if (tile >= nTiles) return;
  const int m = lane & 15, hi = lane >> 4, nl = lane & 15;
  int node = tile * 16 + m;
  int nc = node < N ? node : N - 1;

  v16h ax0 = load_A_g(x + (size_t)nc * 64, 0, hi);
  v16h ax1 = load_A_g(x + (size_t)nc * 64, 1, hi);
  v16h ah0 = load_A_g(h + (size_t)nc * 64, 0, hi);
  v16h ah1 = load_A_g(h + (size_t)nc * 64, 1, hi);

#pragma unroll
  for (int j = 0; j < 4; ++j) {          // 16-wide column chunk of state dim
    v8f ir = {}, iz = {}, inn = {}, hr = {}, hz = {}, hn = {};
    ir  = wmma16(ax0, load_B(Wiht, 192, lane, 0, j),      ir);
    ir  = wmma16(ax1, load_B(Wiht, 192, lane, 1, j),      ir);
    iz  = wmma16(ax0, load_B(Wiht, 192, lane, 0, j + 4),  iz);
    iz  = wmma16(ax1, load_B(Wiht, 192, lane, 1, j + 4),  iz);
    inn = wmma16(ax0, load_B(Wiht, 192, lane, 0, j + 8),  inn);
    inn = wmma16(ax1, load_B(Wiht, 192, lane, 1, j + 8),  inn);
    hr  = wmma16(ah0, load_B(Whht, 192, lane, 0, j),      hr);
    hr  = wmma16(ah1, load_B(Whht, 192, lane, 1, j),      hr);
    hz  = wmma16(ah0, load_B(Whht, 192, lane, 0, j + 4),  hz);
    hz  = wmma16(ah1, load_B(Whht, 192, lane, 1, j + 4),  hz);
    hn  = wmma16(ah0, load_B(Whht, 192, lane, 0, j + 8),  hn);
    hn  = wmma16(ah1, load_B(Whht, 192, lane, 1, j + 8),  hn);

    int n = j * 16 + nl;
    float bir = b_ih[n], biz = b_ih[64 + n], bin = b_ih[128 + n];
    float bhr = b_hh[n], bhz = b_hh[64 + n], bhn = b_hh[128 + n];
#pragma unroll
    for (int v = 0; v < 8; ++v) {
      int nd = tile * 16 + v + 8 * hi;
      float r  = 1.f / (1.f + __expf(-(ir[v] + bir + hr[v] + bhr)));
      float z  = 1.f / (1.f + __expf(-(iz[v] + biz + hz[v] + bhz)));
      float nn = tanhf(inn[v] + bin + r * (hn[v] + bhn));
      if (nd < N) {
        float hold = h[(size_t)nd * 64 + n];
        h[(size_t)nd * 64 + n] = (1.f - z) * nn + z * hold;
      }
    }
  }
}

// ---------------- readout: 64 ->128 relu ->128 relu ->2, softmax ----------------
__global__ __launch_bounds__(128) void k_readout(
    const float* __restrict__ h,
    const _Float16* __restrict__ Wr1t, const float* __restrict__ br1,
    const _Float16* __restrict__ Wr2t, const float* __restrict__ br2,
    const float* __restrict__ Wr3, const float* __restrict__ br3,
    float* __restrict__ out, int nTiles, int N) {
  __shared__ _Float16 act1[WAVES][16 * 128];
  __shared__ _Float16 act2[WAVES][16 * 128];
  const int wave = threadIdx.x >> 5;
  const int lane = threadIdx.x & 31;
  const int tile = blockIdx.x * WAVES + wave;
  if (tile >= nTiles) return;
  const int m = lane & 15, hi = lane >> 4, nl = lane & 15;
  int node = tile * 16 + m;
  int nc = node < N ? node : N - 1;

  v16h a0 = load_A_g(h + (size_t)nc * 64, 0, hi);
  v16h a1 = load_A_g(h + (size_t)nc * 64, 1, hi);

#pragma unroll
  for (int nt = 0; nt < 8; ++nt) {        // layer 1
    v8f acc = {};
    acc = wmma16(a0, load_B(Wr1t, 128, lane, 0, nt), acc);
    acc = wmma16(a1, load_B(Wr1t, 128, lane, 1, nt), acc);
    float bias = br1[nt * 16 + nl];
#pragma unroll
    for (int v = 0; v < 8; ++v) {
      float t = acc[v] + bias;
      act1[wave][(v + 8 * hi) * 128 + nt * 16 + nl] = (_Float16)(t > 0.f ? t : 0.f);
    }
  }
#pragma unroll
  for (int nt = 0; nt < 8; ++nt) {        // layer 2
    v8f acc = {};
#pragma unroll
    for (int kt = 0; kt < 4; ++kt)
      acc = wmma16(load_A_lds(&act1[wave][0], 128, m, hi, kt),
                   load_B(Wr2t, 128, lane, kt, nt), acc);
    float bias = br2[nt * 16 + nl];
#pragma unroll
    for (int v = 0; v < 8; ++v) {
      float t = acc[v] + bias;
      act2[wave][(v + 8 * hi) * 128 + nt * 16 + nl] = (_Float16)(t > 0.f ? t : 0.f);
    }
  }
  // final 128 -> 2 + softmax (lanes 0..15, one node each); vectorized reads
  if (lane < 16) {
    int nd = tile * 16 + lane;
    if (nd < N) {
      const _Float16* a = &act2[wave][lane * 128];
      float o0 = br3[0], o1 = br3[1];
#pragma unroll
      for (int k = 0; k < 128; k += 8) {
        v8h av = *(const v8h*)(a + k);
        float4 w0a = *(const float4*)(Wr3 + k);
        float4 w0b = *(const float4*)(Wr3 + k + 4);
        float4 w1a = *(const float4*)(Wr3 + 128 + k);
        float4 w1b = *(const float4*)(Wr3 + 128 + k + 4);
        o0 += (float)av[0] * w0a.x + (float)av[1] * w0a.y +
              (float)av[2] * w0a.z + (float)av[3] * w0a.w +
              (float)av[4] * w0b.x + (float)av[5] * w0b.y +
              (float)av[6] * w0b.z + (float)av[7] * w0b.w;
        o1 += (float)av[0] * w1a.x + (float)av[1] * w1a.y +
              (float)av[2] * w1a.z + (float)av[3] * w1a.w +
              (float)av[4] * w1b.x + (float)av[5] * w1b.y +
              (float)av[6] * w1b.z + (float)av[7] * w1b.w;
      }
      float mx = fmaxf(o0, o1);
      float e0 = __expf(o0 - mx), e1 = __expf(o1 - mx);
      float s = e0 + e1;
      out[(size_t)nd * 2 + 0] = e0 / s;
      out[(size_t)nd * 2 + 1] = e1 / s;
    }
  }
}

// ---------------- host ----------------

extern "C" void kernel_launch(void* const* d_in, const int* in_sizes, int n_in,
                              void* d_out, int out_size, void* d_ws, size_t ws_size,
                              hipStream_t stream) {
  const float* J    = (const float*)d_in[0];
  const float* b    = (const float*)d_in[1];
  const int*   row  = (const int*)d_in[2];
  const int*   col  = (const int*)d_in[3];
  const float* W1   = (const float*)d_in[4];  const float* b1  = (const float*)d_in[5];
  const float* W2   = (const float*)d_in[6];  const float* b2  = (const float*)d_in[7];
  const float* W3   = (const float*)d_in[8];  const float* b3  = (const float*)d_in[9];
  const float* W_ih = (const float*)d_in[10]; const float* bih = (const float*)d_in[11];
  const float* W_hh = (const float*)d_in[12]; const float* bhh = (const float*)d_in[13];
  const float* Wr1  = (const float*)d_in[14]; const float* br1 = (const float*)d_in[15];
  const float* Wr2  = (const float*)d_in[16]; const float* br2 = (const float*)d_in[17];
  const float* Wr3  = (const float*)d_in[18]; const float* br3 = (const float*)d_in[19];

  const int N = in_sizes[1];       // 5000
  const int E = in_sizes[2];       // 80000

  size_t o = 0;
  auto take = [&](size_t bytes) {
    o = (o + 255) & ~(size_t)255;
    void* p = (char*)d_ws + o;
    o += bytes;
    return p;
  };
  float*    hbuf = (float*)take((size_t)N * 64 * 4);
  float*    xbuf = (float*)take((size_t)N * 64 * 4);
  float*    ef   = (float*)take((size_t)E * 4 * 4);
  _Float16* W1t  = (_Float16*)take(160 * 128 * 2);
  _Float16* W2t  = (_Float16*)take(128 * 128 * 2);
  _Float16* W3t  = (_Float16*)take(128 * 64 * 2);
  _Float16* Wiht = (_Float16*)take(64 * 192 * 2);
  _Float16* Whht = (_Float16*)take(64 * 192 * 2);
  _Float16* Wr1t = (_Float16*)take(64 * 128 * 2);
  _Float16* Wr2t = (_Float16*)take(128 * 128 * 2);
  (void)ws_size;

  // ---- prep
  k_zero<<<(N * 64 + 255) / 256, 256, 0, stream>>>(hbuf, N * 64);
  k_transpose_cvt<<<(160 * 128 + 255) / 256, 256, 0, stream>>>(W1, W1t, 128, 132, 160);
  k_transpose_cvt<<<(128 * 128 + 255) / 256, 256, 0, stream>>>(W2, W2t, 128, 128, 128);
  k_transpose_cvt<<<(128 * 64  + 255) / 256, 256, 0, stream>>>(W3, W3t, 64, 128, 128);
  k_transpose_cvt<<<(64 * 192  + 255) / 256, 256, 0, stream>>>(W_ih, Wiht, 192, 64, 64);
  k_transpose_cvt<<<(64 * 192  + 255) / 256, 256, 0, stream>>>(W_hh, Whht, 192, 64, 64);
  k_transpose_cvt<<<(64 * 128  + 255) / 256, 256, 0, stream>>>(Wr1, Wr1t, 128, 64, 64);
  k_transpose_cvt<<<(128 * 128 + 255) / 256, 256, 0, stream>>>(Wr2, Wr2t, 128, 128, 128);
  k_edge_feat<<<(E + 255) / 256, 256, 0, stream>>>(J, b, row, col, ef, E, N);

  const int eTiles = (E + 15) / 16;
  const int eBlocks = (eTiles + WAVES - 1) / WAVES;
  const int nTiles = (N + 15) / 16;
  const int nBlocks = (nTiles + WAVES - 1) / WAVES;

  for (int step = 0; step < 10; ++step) {
    k_zero<<<(N * 64 + 255) / 256, 256, 0, stream>>>(xbuf, N * 64);
    k_edge_mlp<<<eBlocks, 128, 0, stream>>>(hbuf, ef, row, col,
                                            W1t, b1, W2t, b2, W3t, b3,
                                            xbuf, E, eTiles);
    k_gru<<<nBlocks, 128, 0, stream>>>(xbuf, hbuf, Wiht, bih, Whht, bhh, nTiles, N);
  }
  k_readout<<<nBlocks, 128, 0, stream>>>(hbuf, Wr1t, br1, Wr2t, br2, Wr3, br3,
                                         (float*)d_out, nTiles, N);
}
